// UserTower_36515811950887
// MI455X (gfx1250) — matrix operand order repaired and checked by reading, not compiled, not run on hardware
//
#include <hip/hip_runtime.h>
#include <hip/hip_bf16.h>
#include <cstdint>

// ---------------- problem constants ----------------
#define EMB       256
#define HIDDEN    1024
#define HEADS     8
#define HEAD_DIM  32
#define BATCH     2048
#define SEQ       200
#define SEQ_PAD   224                 // 7 chunks of 32
#define NCHUNK    7
#define ROWS      (BATCH * SEQ)       // 409600 gathered item rows

typedef __attribute__((ext_vector_type(16))) __bf16 v16bf;
typedef __attribute__((ext_vector_type(8)))  __bf16 v8bf;
typedef __attribute__((ext_vector_type(4)))  __bf16 v4bf;
typedef __attribute__((ext_vector_type(8)))  float  v8f;

// ---------------- WMMA helper (CDNA5 16x16x32 bf16 -> f32) ----------------
__device__ __forceinline__ v8f wmma_bf16(v16bf a, v16bf b, v8f c) {
  return __builtin_amdgcn_wmma_f32_16x16x32_bf16(false, a, false, b, (short)0, c,
                                                 false, false);
}

// A fragment from LDS tile (row-major, configurable row stride in elements).
__device__ __forceinline__ v16bf load_a(const __bf16* As, int stride, int mi, int lane) {
  const int m    = (lane & 15) + mi * 16;
  const int koff = (lane >> 4) * 8;
  const __bf16* row = As + m * stride;
  v8bf lo = *(const v8bf*)(row + koff);
  v8bf hi = *(const v8bf*)(row + koff + 16);
  return __builtin_shufflevector(lo, hi, 0,1,2,3,4,5,6,7,8,9,10,11,12,13,14,15);
}

// B fragment from pre-swizzled LDS tile: each lane reads 16 contiguous bf16 (32B).
__device__ __forceinline__ v16bf load_b(const __bf16* Bs, int ct, int lane) {
  const __bf16* p = Bs + ct * 512 + (lane >> 4) * 256 + (lane & 15) * 16;
  return *(const v16bf*)p;
}

// swizzled index for B element (k in [0,32), n in [0, 16*tiles))
__device__ __host__ __forceinline__ int bswz(int k, int n) {
  return (n >> 4) * 512 + ((k >> 3) & 1) * 256 + (n & 15) * 16 + ((k >> 4) << 3) + (k & 7);
}

// ---------------- CDNA5 async global->LDS (ASYNCcnt-tracked DMA) ----------------
__device__ __forceinline__ void async_g2l_b128(uint32_t lds_byte_addr,
                                               unsigned long long gbase,
                                               uint32_t goff) {
  asm volatile("global_load_async_to_lds_b128 %0, %1, %2"
               :: "v"(lds_byte_addr), "v"(goff), "s"(gbase)
               : "memory");
}
__device__ __forceinline__ void wait_asynccnt0() {
  asm volatile("s_wait_asynccnt 0x0" ::: "memory");
}
__device__ __forceinline__ void wait_asynccnt4() {   // keep next tile's loads in flight
  asm volatile("s_wait_asynccnt 0x4" ::: "memory");
}
__device__ __forceinline__ uint32_t lds_addr32(const void* p) {
  return (uint32_t)(uintptr_t)p;   // LDS aperture: low 32 bits == LDS byte address
}

__device__ __forceinline__ v4bf cvt4(float4 v) {
  v4bf b;
  b[0] = (__bf16)v.x; b[1] = (__bf16)v.y; b[2] = (__bf16)v.z; b[3] = (__bf16)v.w;
  return b;
}

// ---------------- weight packing: fp32 -> bf16, blocked+swizzled ----------------
// dst layout: [K/32][N/256] blocks of 8192 bf16 (the exact 16KB LDS image).
__global__ __launch_bounds__(256) void pack_wswz_k(const float* __restrict__ src,
                                                   __bf16* __restrict__ dst,
                                                   int K, int N) {
  int i = blockIdx.x * 256 + threadIdx.x;
  if (i >= K * N) return;
  int k = i / N, n = i % N;
  int nblk = N >> 8;
  size_t blk = (size_t)((k >> 5) * nblk + (n >> 8)) << 13;
  dst[blk + bswz(k & 31, n & 255)] = (__bf16)src[i];
}

// Wvk blocked: [8 k-steps] x 8704 bf16 (17 tiles: 16 = Wv cols, 1 = Wk head-reduction).
__global__ __launch_bounds__(256) void pack_wvk_k(const float* __restrict__ Wv,
                                                  const float* __restrict__ Wk,
                                                  const float* __restrict__ bk,
                                                  __bf16* __restrict__ Wvk,
                                                  float* __restrict__ bkred) {
  int i = blockIdx.x * 256 + threadIdx.x;
  if (i < HEADS) {
    float s = 0.f;
    for (int d = 0; d < HEAD_DIM; ++d) s += bk[i * HEAD_DIM + d];
    bkred[i] = s;
  }
  if (i < EMB * 272) {
    int k = i / 272, n = i % 272;
    __bf16 v;
    if (n < EMB) {
      v = (__bf16)Wv[k * EMB + n];
    } else if (n < EMB + HEADS) {
      int h = n - EMB;
      float s = 0.f;
      for (int d = 0; d < HEAD_DIM; ++d) s += Wk[k * EMB + h * HEAD_DIM + d];
      v = (__bf16)s;
    } else {
      v = (__bf16)0.f;
    }
    Wvk[(size_t)(k >> 5) * 8704 + bswz(k & 31, n)] = v;
  }
}

// zero the n=200..223 padding rows of blocked V (chunk 6, k=8..31)
__global__ __launch_bounds__(256) void zero_vpad_k(__bf16* __restrict__ Vb) {
  int i = blockIdx.x * 256 + threadIdx.x;
  if (i >= BATCH * 24 * 256) return;
  int b = i / (24 * 256);
  int rem = i % (24 * 256);
  int k = 8 + rem / 256;
  int col = rem % 256;
  Vb[(((size_t)b * NCHUNK + 6) << 13) + bswz(k, col)] = (__bf16)0.f;
}

// ---------------- gathered GEMM: V = it@Wv + bv (bf16 blocked), ks = it@Wk_red + bk_red ----------------
__global__ __launch_bounds__(256)
void gather_v_ks_k(const float* __restrict__ item_table, const int* __restrict__ item_ids,
                   const __bf16* __restrict__ Wvk, const float* __restrict__ bv,
                   const float* __restrict__ bkred,
                   __bf16* __restrict__ Vb, float* __restrict__ ks) {
  __shared__ __attribute__((aligned(32))) __bf16 As[32 * 264];  // full 32x256 A tile (+pad)
  __shared__ __attribute__((aligned(32))) __bf16 Bs[2][17 * 512];
  const int tid = threadIdx.x, lane = tid & 31, wave = tid >> 5;
  const int rowbase = blockIdx.x * 32;

  auto issueB = [&](int kb, int buf) {
    unsigned long long src = (unsigned long long)(uintptr_t)(Wvk + (size_t)kb * 8704);
    uint32_t dst = lds_addr32(&Bs[buf][0]);
    for (uint32_t i = (uint32_t)tid * 16u; i < 17408u; i += 4096u)
      async_g2l_b128(dst + i, src, i);
  };

  issueB(0, 0);

  // stage entire gathered A tile up front: 8 threads per row, 8 independent b128 loads each
  {
    const int r = tid >> 3, c4 = tid & 7;
    const int ar = item_ids[rowbase + r];
    const float4* rowp = (const float4*)(item_table + (size_t)ar * EMB);
    float4 v[8];
#pragma unroll
    for (int j = 0; j < 8; ++j) v[j] = rowp[j * 8 + c4];   // all loads in flight
#pragma unroll
    for (int j = 0; j < 8; ++j)
      *(v4bf*)&As[r * 264 + j * 32 + c4 * 4] = cvt4(v[j]);
  }

  v8f acc[2][2], ak[2];
#pragma unroll
  for (int mi = 0; mi < 2; ++mi) {
#pragma unroll
    for (int ni = 0; ni < 2; ++ni)
#pragma unroll
      for (int r = 0; r < 8; ++r) acc[mi][ni][r] = 0.f;
#pragma unroll
    for (int r = 0; r < 8; ++r) ak[mi][r] = 0.f;
  }

  for (int kb = 0; kb < EMB / 32; ++kb) {
    const int cur = kb & 1;
    if (kb + 1 < EMB / 32) {
      issueB(kb + 1, cur ^ 1);
      wait_asynccnt4();      // current tile done (in-order), next stays in flight
    } else {
      wait_asynccnt0();
    }
    __syncthreads();

    v16bf a0 = load_a(As + kb * 32, 264, 0, lane);
    v16bf a1 = load_a(As + kb * 32, 264, 1, lane);
#pragma unroll
    for (int ni = 0; ni < 2; ++ni) {
      v16bf b = load_b(Bs[cur], wave * 2 + ni, lane);
      acc[0][ni] = wmma_bf16(a0, b, acc[0][ni]);
      acc[1][ni] = wmma_bf16(a1, b, acc[1][ni]);
    }
    if (wave == 0) {  // wave-uniform branch; EXEC all ones inside
      v16bf bk_ = load_b(Bs[cur], 16, lane);
      ak[0] = wmma_bf16(a0, bk_, ak[0]);
      ak[1] = wmma_bf16(a1, bk_, ak[1]);
    }
    __syncthreads();
  }

  // epilogue: V in blocked-swizzled layout [b][chunk][8192]
#pragma unroll
  for (int mi = 0; mi < 2; ++mi)
#pragma unroll
    for (int ni = 0; ni < 2; ++ni)
#pragma unroll
      for (int r = 0; r < 8; ++r) {
        int row = rowbase + mi * 16 + (lane >> 4) * 8 + r;
        int b = row / SEQ, n = row % SEQ;
        int col = (wave * 2 + ni) * 16 + (lane & 15);
        Vb[(((size_t)b * NCHUNK + (n >> 5)) << 13) + bswz(n & 31, col)] =
            (__bf16)(acc[mi][ni][r] + bv[col]);
      }
  if (wave == 0) {
#pragma unroll
    for (int mi = 0; mi < 2; ++mi)
#pragma unroll
      for (int r = 0; r < 8; ++r) {
        int row = rowbase + mi * 16 + (lane >> 4) * 8 + r;
        int col = lane & 15;
        if (col < HEADS) ks[(size_t)row * HEADS + col] = ak[mi][r] + bkred[col];
      }
  }
}

// ---------------- generic WMMA GEMM: out = [relu]( [out +] A@Bw + bias*scale ) ----------------
// Bw is blocked+swizzled: [K/32][nblk] x 8192 bf16. Double-buffered async DMA.
template <bool GATHER, bool ACC, bool RELU>
__global__ __launch_bounds__(256)
void gemm_bf16wmma_k(const float* __restrict__ A, const int* __restrict__ idx, int lda,
                     const __bf16* __restrict__ Bw, int nblk,
                     const float* __restrict__ bias, float bscale,
                     float* __restrict__ out, int ldo, int K) {
  __shared__ __attribute__((aligned(32))) __bf16 As[32 * 40];
  __shared__ __attribute__((aligned(32))) __bf16 Bs[2][16 * 512];
  const int tid = threadIdx.x, lane = tid & 31, wave = tid >> 5;
  const int rowbase = blockIdx.x * 32;
  const int colbase = blockIdx.y * 256;
  const int nk = K >> 5;

  // per-thread A staging assignment: 8 threads per row, one float4 chunk per step
  const int r8 = tid >> 3, c4 = tid & 7;
  const int arow = rowbase + r8;
  const int ar = GATHER ? idx[arow] : arow;
  const float4* rowp = (const float4*)(A + (size_t)ar * lda);

  v8f acc[2][2];
#pragma unroll
  for (int mi = 0; mi < 2; ++mi)
#pragma unroll
    for (int ni = 0; ni < 2; ++ni)
#pragma unroll
      for (int r = 0; r < 8; ++r) acc[mi][ni][r] = 0.f;

  auto issueB = [&](int kb, int buf) {
    unsigned long long src =
        (unsigned long long)(uintptr_t)(Bw + ((size_t)(kb * nblk + blockIdx.y) << 13));
    uint32_t dst = lds_addr32(&Bs[buf][0]) + (uint32_t)tid * 16u;
#pragma unroll
    for (int j = 0; j < 4; ++j)
      async_g2l_b128(dst + j * 4096u, src, (uint32_t)tid * 16u + j * 4096u);
  };

  issueB(0, 0);
  for (int kb = 0; kb < nk; ++kb) {
    const int cur = kb & 1;
    if (kb + 1 < nk) issueB(kb + 1, cur ^ 1);
    // stage A tile chunk (fp32 -> bf16), one b128 load + b64 LDS store per thread
    float4 av = rowp[kb * 8 + c4];
    *(v4bf*)&As[r8 * 40 + c4 * 4] = cvt4(av);
    if (kb + 1 < nk) {
      __builtin_prefetch(&rowp[(kb + 1) * 8 + c4], 0, 1);
      wait_asynccnt4();
    } else {
      wait_asynccnt0();
    }
    __syncthreads();

    v16bf a0 = load_a(As, 40, 0, lane);
    v16bf a1 = load_a(As, 40, 1, lane);
#pragma unroll
    for (int ni = 0; ni < 2; ++ni) {
      v16bf b = load_b(Bs[cur], wave * 2 + ni, lane);
      acc[0][ni] = wmma_bf16(a0, b, acc[0][ni]);
      acc[1][ni] = wmma_bf16(a1, b, acc[1][ni]);
    }
    __syncthreads();
  }

#pragma unroll
  for (int mi = 0; mi < 2; ++mi)
#pragma unroll
    for (int ni = 0; ni < 2; ++ni)
#pragma unroll
      for (int r = 0; r < 8; ++r) {
        int row = rowbase + mi * 16 + (lane >> 4) * 8 + r;
        int col = colbase + (wave * 2 + ni) * 16 + (lane & 15);
        float v = acc[mi][ni][r];
        if (bias) v += bias[col] * bscale;
        if (ACC) v += out[(size_t)row * ldo + col];
        if (RELU) v = fmaxf(v, 0.f);
        out[(size_t)row * ldo + col] = v;
      }
}

// ---------------- attention: per-batch block, wave = head ----------------
__global__ __launch_bounds__(256)
void attention_k(const float* __restrict__ q_ws, const float* __restrict__ ks_ws,
                 const __bf16* __restrict__ Vb, float* __restrict__ S_ws) {
  __shared__ __attribute__((aligned(16))) float ks_l[SEQ_PAD * HEADS];
  __shared__ float q_l[EMB];
  __shared__ __attribute__((aligned(32))) __bf16 Vs[2][16 * 512];
  __shared__ float Sp[HEADS * EMB];

  const int b = blockIdx.x;
  const int tid = threadIdx.x, lane = tid & 31, h = tid >> 5;

  auto issueV = [&](int c, int buf) {
    unsigned long long src =
        (unsigned long long)(uintptr_t)(Vb + (((size_t)b * NCHUNK + c) << 13));
    uint32_t dst = lds_addr32(&Vs[buf][0]) + (uint32_t)tid * 16u;
#pragma unroll
    for (int j = 0; j < 4; ++j)
      async_g2l_b128(dst + j * 4096u, src, (uint32_t)tid * 16u + j * 4096u);
  };

  // async DMA: ks block (200*8 f32 = 6400B linear; pad region never read) + first V chunk
  {
    unsigned long long src = (unsigned long long)(uintptr_t)(ks_ws + (size_t)b * SEQ * HEADS);
    uint32_t dst = lds_addr32(&ks_l[0]);
    for (uint32_t i = (uint32_t)tid * 16u; i < (uint32_t)SEQ * HEADS * 4u; i += 4096u)
      async_g2l_b128(dst + i, src, i);
  }
  issueV(0, 0);
  q_l[tid] = q_ws[(size_t)b * EMB + tid];
  wait_asynccnt0();
  __syncthreads();

  // per-head min/max of ks (softmax of scalar*vector)
  float mx = -1e30f, mn = 1e30f;
  for (int n = lane; n < SEQ; n += 32) {
    float v = ks_l[n * HEADS + h];
    mx = fmaxf(mx, v);
    mn = fminf(mn, v);
  }
#pragma unroll
  for (int o = 16; o >= 1; o >>= 1) {
    mx = fmaxf(mx, __shfl_xor(mx, o, 32));
    mn = fminf(mn, __shfl_xor(mn, o, 32));
  }

  const float inv_s = 0.17677669529663687f;  // 1/sqrt(HEAD_DIM)
  float qv = q_l[h * HEAD_DIM + lane];       // each lane owns one q-row
  float rm = (qv >= 0.f) ? qv * mx : qv * mn;
  float ssum = 0.f;
  for (int n = 0; n < SEQ; ++n) ssum += __expf((qv * ks_l[n * HEADS + h] - rm) * inv_s);
  float inv = 1.f / ssum;

  v8f acc[2][2];
#pragma unroll
  for (int mi = 0; mi < 2; ++mi)
#pragma unroll
    for (int ni = 0; ni < 2; ++ni)
#pragma unroll
      for (int r = 0; r < 8; ++r) acc[mi][ni][r] = 0.f;

  const int koff = (lane >> 4) * 8;
  for (int c = 0; c < NCHUNK; ++c) {
    const int cur = c & 1;
    if (c + 1 < NCHUNK) {
      issueV(c + 1, cur ^ 1);
      wait_asynccnt4();
    } else {
      wait_asynccnt0();
    }
    __syncthreads();

    // build softmax A-fragments directly in registers
    v16bf am[2];
#pragma unroll
    for (int mi = 0; mi < 2; ++mi) {
      int src = (lane & 15) + mi * 16;
      float qv_m = __shfl(qv, src, 32);
      float rm_m = __shfl(rm, src, 32);
      float in_m = __shfl(inv, src, 32);
#pragma unroll
      for (int e = 0; e < 16; ++e) {
        int kk = koff + (e < 8 ? e : e + 8);
        int n = c * 32 + kk;
        float w = 0.f;
        if (n < SEQ) w = __expf((qv_m * ks_l[n * HEADS + h] - rm_m) * inv_s) * in_m;
        am[mi][e] = (__bf16)w;
      }
    }
#pragma unroll
    for (int ni = 0; ni < 2; ++ni) {
      v16bf bfrag = load_b(Vs[cur], h * 2 + ni, lane);
      acc[0][ni] = wmma_bf16(am[0], bfrag, acc[0][ni]);
      acc[1][ni] = wmma_bf16(am[1], bfrag, acc[1][ni]);
    }
    __syncthreads();
  }

  // fold out[h,q,k] into S[b,(q&7)*32+k]; q&7 == r in this C layout.
  float p[8][2];
#pragma unroll
  for (int r = 0; r < 8; ++r)
#pragma unroll
    for (int ni = 0; ni < 2; ++ni) p[r][ni] = acc[0][ni][r] + acc[1][ni][r];
#pragma unroll
  for (int r = 0; r < 8; ++r)
#pragma unroll
    for (int ni = 0; ni < 2; ++ni) {
      float v = p[r][ni] + __shfl_xor(p[r][ni], 16, 32);
      if (lane < 16) Sp[h * EMB + r * 32 + ni * 16 + lane] = v;
    }
  __syncthreads();
  {
    float s = 0.f;
#pragma unroll
    for (int hh = 0; hh < HEADS; ++hh) s += Sp[hh * EMB + tid];
    S_ws[(size_t)b * EMB + tid] = s;
  }
}

// ---------------- host launch ----------------
extern "C" void kernel_launch(void* const* d_in, const int* in_sizes, int n_in,
                              void* d_out, int out_size, void* d_ws, size_t ws_size,
                              hipStream_t stream) {
  const int*   user_id    = (const int*)d_in[0];
  const int*   item_ids   = (const int*)d_in[1];
  const float* user_table = (const float*)d_in[2];
  const float* item_table = (const float*)d_in[3];
  const float* Wq = (const float*)d_in[4];
  const float* bq = (const float*)d_in[5];
  const float* Wk = (const float*)d_in[6];
  const float* bk = (const float*)d_in[7];
  const float* Wv = (const float*)d_in[8];
  const float* bv = (const float*)d_in[9];
  const float* Wo = (const float*)d_in[10];
  const float* bo = (const float*)d_in[11];
  const float* W1 = (const float*)d_in[12];
  const float* b1 = (const float*)d_in[13];
  const float* W2 = (const float*)d_in[14];
  const float* b2 = (const float*)d_in[15];
  float* out = (float*)d_out;

  char* w = (char*)d_ws;
  auto alloc = [&](size_t bytes) -> void* {
    void* p = (void*)w;
    w += (bytes + 255) & ~(size_t)255;
    return p;
  };
  __bf16* Wq_bf  = (__bf16*)alloc((size_t)EMB * EMB * 2);
  __bf16* Wo_bf  = (__bf16*)alloc((size_t)EMB * EMB * 2);
  __bf16* W1_bf  = (__bf16*)alloc((size_t)2 * EMB * HIDDEN * 2);
  __bf16* W2_bf  = (__bf16*)alloc((size_t)HIDDEN * EMB * 2);
  __bf16* Wvk_bf = (__bf16*)alloc((size_t)8 * 8704 * 2);
  float*  bkred  = (float*)alloc(HEADS * 4);
  __bf16* Vb     = (__bf16*)alloc((size_t)BATCH * NCHUNK * 8192 * 2);
  float*  ks_ws  = (float*)alloc((size_t)ROWS * HEADS * 4);
  float*  q_ws   = (float*)alloc((size_t)BATCH * EMB * 4);
  float*  S_ws   = (float*)alloc((size_t)BATCH * EMB * 4);
  float*  ur_ws  = (float*)alloc((size_t)BATCH * EMB * 4);
  float*  h1_ws  = (float*)alloc((size_t)BATCH * HIDDEN * 4);

  // 1) weight prep (bf16 + blocked/swizzled so GEMM B-tiles are linear 16KB DMAs)
  pack_wswz_k<<<(EMB * EMB + 255) / 256, 256, 0, stream>>>(Wq, Wq_bf, EMB, EMB);
  pack_wswz_k<<<(EMB * EMB + 255) / 256, 256, 0, stream>>>(Wo, Wo_bf, EMB, EMB);
  pack_wswz_k<<<(2 * EMB * HIDDEN + 255) / 256, 256, 0, stream>>>(W1, W1_bf, 2 * EMB, HIDDEN);
  pack_wswz_k<<<(HIDDEN * EMB + 255) / 256, 256, 0, stream>>>(W2, W2_bf, HIDDEN, EMB);
  pack_wvk_k<<<(EMB * 272 + 255) / 256, 256, 0, stream>>>(Wv, Wk, bk, Wvk_bf, bkred);
  zero_vpad_k<<<(BATCH * 24 * 256 + 255) / 256, 256, 0, stream>>>(Vb);

  // 2) gathered big GEMM: V (bf16 blocked) + ks (f32)
  gather_v_ks_k<<<ROWS / 32, 256, 0, stream>>>(item_table, item_ids, Wvk_bf, bv, bkred, Vb, ks_ws);

  // 3) q = gather(u) @ Wq + bq
  gemm_bf16wmma_k<true, false, false><<<dim3(BATCH / 32, 1), 256, 0, stream>>>(
      user_table, user_id, EMB, Wq_bf, 1, bq, 1.0f, q_ws, EMB, EMB);

  // 4) attention -> S[b,256]
  attention_k<<<BATCH, 256, 0, stream>>>(q_ws, ks_ws, Vb, S_ws);

  // 5) user_rep = S @ Wo + 32*bo
  gemm_bf16wmma_k<false, false, false><<<dim3(BATCH / 32, 1), 256, 0, stream>>>(
      S_ws, nullptr, EMB, Wo_bf, 1, bo, 32.0f, ur_ws, EMB, EMB);

  // 6) h1 = relu( u @ W1[0:256] + user_rep @ W1[256:512] + b1 )
  gemm_bf16wmma_k<true, false, false><<<dim3(BATCH / 32, HIDDEN / 256), 256, 0, stream>>>(
      user_table, user_id, EMB, W1_bf, HIDDEN / 256, b1, 1.0f, h1_ws, HIDDEN, EMB);
  gemm_bf16wmma_k<false, true, true><<<dim3(BATCH / 32, HIDDEN / 256), 256, 0, stream>>>(
      ur_ws, nullptr, EMB, W1_bf + (size_t)8 * (HIDDEN / 256) * 8192, HIDDEN / 256,
      nullptr, 0.0f, h1_ws, HIDDEN, EMB);

  // 7) out = h1 @ W2 + b2
  gemm_bf16wmma_k<false, false, false><<<dim3(BATCH / 32, 1), 256, 0, stream>>>(
      h1_ws, nullptr, HIDDEN, W2_bf, 1, b2, 1.0f, out, EMB, HIDDEN);
}
